// LSHEncoderLayer_15341623181662
// MI455X (gfx1250) — matrix operand order repaired and checked by reading, not compiled, not run on hardware
//
#include <hip/hip_runtime.h>
#include <hip/hip_bf16.h>

// ---------------------------------------------------------------------------
// LSH encoder layer for MI455X (gfx1250), bf16 WMMA everywhere.
// Shapes fixed by the reference: B=8 S=4096 D=512 H=8 Dh=64 R=4 BUCKET=64
// n_buckets=64 Dff=2048.
// ---------------------------------------------------------------------------

typedef __attribute__((ext_vector_type(16))) __bf16 bf16x16;
typedef __attribute__((ext_vector_type(8)))  float  f32x8;

union BF16Frag { bf16x16 v; unsigned short u[16]; };

__device__ __forceinline__ unsigned short f2bf(float f) {
    unsigned int u = __float_as_uint(f);
    unsigned int r = ((u >> 16) & 1u) + 0x7FFFu;   // round-to-nearest-even
    return (unsigned short)((u + r) >> 16);
}

__device__ __forceinline__ f32x8 zero8() {
    f32x8 z;
#pragma unroll
    for (int i = 0; i < 8; ++i) z[i] = 0.0f;
    return z;
}

__device__ __forceinline__ f32x8 wmma_bf16(bf16x16 a, bf16x16 b, f32x8 c) {
    // 8 args: (neg_a, A, neg_b, B, c_mod, C, reuse_a, reuse_b)
    return __builtin_amdgcn_wmma_f32_16x16x32_bf16(false, a, false, b,
                                                   (short)0, c, false, false);
}

// A fragment (16x32 bf16) from row-major LDS tile [m][k], row stride `ld`.
// Contiguous k order (matches load_frag_B): lane = khalf*16 + m; u[j] = K(khalf*16 + j)
__device__ __forceinline__ bf16x16 load_frag_A(const unsigned short* smem, int ld,
                                               int mBase, int kBase, int lane) {
    int m  = mBase + (lane & 15);
    int kh = (lane >> 4) * 16;
    const unsigned short* p0 = smem + m * ld + kBase + kh;
    BF16Frag f;
#pragma unroll
    for (int j = 0; j < 16; ++j) f.u[j] = p0[j];
    return f.v;
}

// B fragment (32x16 bf16) from LDS stored TRANSPOSED as [n][k], row stride `ld`.
// Contiguous k order (matches load_frag_A): lane = khalf*16 + n; u[j] = K(khalf*16 + j)
__device__ __forceinline__ bf16x16 load_frag_B(const unsigned short* smem, int ld,
                                               int nBase, int kBase, int lane) {
    int n = nBase + (lane & 15);
    int k = kBase + (lane >> 4) * 16;
    const unsigned short* p = smem + n * ld + k;
    BF16Frag f;
#pragma unroll
    for (int j = 0; j < 16; ++j) f.u[j] = p[j];
    return f.v;
}

// staging-store helpers (convert fp32 -> bf16; pass through bf16)
__device__ __forceinline__ void stage_storeA(unsigned short* d, const float4& f) {
    d[0] = f2bf(f.x); d[1] = f2bf(f.y); d[2] = f2bf(f.z); d[3] = f2bf(f.w);
}
__device__ __forceinline__ void stage_storeA(unsigned short* d, const ushort4& u) {
    *(ushort4*)d = u;
}

template <typename T> struct Vec4;
template <> struct Vec4<float>          { using type = float4;  };
template <> struct Vec4<unsigned short> { using type = ushort4; };

// ---------------------------------------------------------------------------
// Double-buffered GEMM: C[M,N] = act(A[M,K] @ B[K,N] + bias)
// AT = float (converted to bf16 on stage) or unsigned short (bf16 bits).
// B: fp32 weights (bf16-converted on stage, transposed in LDS).
// Tile 128x128x32, 256 threads = 8 waves (4x2), each wave 32x64 (2x4 wmma).
// Ping-pong LDS: fetch tile k+1 into regs before computing tile k, stage after,
// single barrier per K-step -> global latency overlapped with WMMA stream.
// M,N multiples of 128; K multiple of 32 (true for all shapes here).
// ---------------------------------------------------------------------------
template <typename AT>
__global__ __launch_bounds__(256) void gemm_dbuf_kernel(
    const AT* __restrict__ A,
    const float* __restrict__ B,
    const float* __restrict__ bias,
    void* __restrict__ Cv, int cIsBf16, int doRelu,
    int M, int N, int K)
{
    __shared__ unsigned short As[2][128 * 32];   // [m][k]
    __shared__ unsigned short Bs[2][128 * 32];   // transposed: [n][k]

    const int tid  = threadIdx.x;
    const int lane = tid & 31;
    const int wave = tid >> 5;
    const int wm   = wave >> 1;               // 0..3 -> rows wm*32
    const int wn   = wave & 1;                // 0..1 -> cols wn*64
    const int m0   = blockIdx.x * 128;
    const int n0   = blockIdx.y * 128;
    const int nk   = K >> 5;

    // per-thread staging coordinates
    const int arow = tid >> 3;                // +32*p : 128 rows / 4 passes
    const int ac4  = (tid & 7) << 2;          // 4 of 32 k-columns
    const int bkr  = tid >> 5;                // +8*p  : 32 k-rows / 4 passes
    const int bc4  = (tid & 31) << 2;         // 4 of 128 n-columns

    typename Vec4<AT>::type aReg[4];
    float4 bReg[4];

    f32x8 acc[2][4];
#pragma unroll
    for (int i = 0; i < 2; ++i)
#pragma unroll
        for (int j = 0; j < 4; ++j) acc[i][j] = zero8();

    auto fetch = [&](int kt) {
        const int k0 = kt << 5;
#pragma unroll
        for (int p = 0; p < 4; ++p) {
            const AT* src = A + (size_t)(m0 + arow + p * 32) * K + k0 + ac4;
            aReg[p] = *(const typename Vec4<AT>::type*)src;
            if (kt + 1 < nk) __builtin_prefetch(src + 32, 0, 3);
        }
#pragma unroll
        for (int p = 0; p < 4; ++p) {
            const float* src = B + (size_t)(k0 + bkr + p * 8) * N + n0 + bc4;
            bReg[p] = *(const float4*)src;
            if (kt + 1 < nk) __builtin_prefetch(src + (size_t)32 * N, 0, 3);
        }
    };

    auto stage = [&](int buf) {
#pragma unroll
        for (int p = 0; p < 4; ++p)
            stage_storeA(&As[buf][(arow + p * 32) * 32 + ac4], aReg[p]);
#pragma unroll
        for (int p = 0; p < 4; ++p) {
            const int kr = bkr + p * 8;
            float4 f = bReg[p];
            Bs[buf][(bc4 + 0) * 32 + kr] = f2bf(f.x);
            Bs[buf][(bc4 + 1) * 32 + kr] = f2bf(f.y);
            Bs[buf][(bc4 + 2) * 32 + kr] = f2bf(f.z);
            Bs[buf][(bc4 + 3) * 32 + kr] = f2bf(f.w);
        }
    };

    fetch(0);
    stage(0);
    __syncthreads();

    for (int kt = 0; kt < nk; ++kt) {
        const int cur = kt & 1;
        if (kt + 1 < nk) fetch(kt + 1);       // loads in flight during compute

        bf16x16 afrag[2], bfrag[4];
#pragma unroll
        for (int mt = 0; mt < 2; ++mt)
            afrag[mt] = load_frag_A(As[cur], 32, wm * 32 + mt * 16, 0, lane);
#pragma unroll
        for (int nt = 0; nt < 4; ++nt)
            bfrag[nt] = load_frag_B(Bs[cur], 32, wn * 64 + nt * 16, 0, lane);
#pragma unroll
        for (int mt = 0; mt < 2; ++mt)
#pragma unroll
            for (int nt = 0; nt < 4; ++nt)
                acc[mt][nt] = wmma_bf16(afrag[mt], bfrag[nt], acc[mt][nt]);

        if (kt + 1 < nk) stage(cur ^ 1);      // writes other buffer: no WAR
        __syncthreads();
    }

    // ---- epilogue: C/D layout m = (lane>>4)*8 + r, n = lane&15 ----
#pragma unroll
    for (int mt = 0; mt < 2; ++mt) {
#pragma unroll
        for (int nt = 0; nt < 4; ++nt) {
            int ng  = n0 + wn * 64 + nt * 16 + (lane & 15);
            int mg0 = m0 + wm * 32 + mt * 16 + ((lane >> 4) << 3);
            float bv = bias ? bias[ng] : 0.0f;
#pragma unroll
            for (int r = 0; r < 8; ++r) {
                float val = acc[mt][nt][r] + bv;
                if (doRelu) val = fmaxf(val, 0.0f);
                size_t off = (size_t)(mg0 + r) * N + ng;
                if (cIsBf16) ((unsigned short*)Cv)[off] = f2bf(val);
                else         ((float*)Cv)[off] = val;
            }
        }
    }
}

// ---------------------------------------------------------------------------
// Hash: sort_key[r,b,h,s] = argmax_k([qk@rot, -qk@rot]) * S + s
// One thread per (r,b,h,s). rot is [R, Dh=64, 32]; r,b,h uniform per wave
// so rot loads are scalar.
// ---------------------------------------------------------------------------
__global__ __launch_bounds__(256) void hash_kernel(
    const float* __restrict__ qk, const float* __restrict__ rot,
    int* __restrict__ key)
{
    int gid = blockIdx.x * 256 + threadIdx.x;      // R*B*H*S = 1048576
    int s = gid & 4095;
    int h = (gid >> 12) & 7;
    int b = (gid >> 15) & 7;
    int r = gid >> 18;
    const float* q  = qk  + ((size_t)(b * 4096 + s)) * 512 + h * 64;
    const float* rt = rot + (size_t)r * 64 * 32;

    float accv[32];
#pragma unroll
    for (int k = 0; k < 32; ++k) accv[k] = 0.0f;
    for (int d = 0; d < 64; ++d) {
        float xd = q[d];
        const float* rr = rt + d * 32;
#pragma unroll
        for (int k = 0; k < 32; ++k) accv[k] = fmaf(xd, rr[k], accv[k]);
    }
    float best = -3.0e38f; int bi = 0;
#pragma unroll
    for (int k = 0; k < 32; ++k) { if (accv[k] > best) { best = accv[k]; bi = k; } }
#pragma unroll
    for (int k = 0; k < 32; ++k) { float v2 = -accv[k]; if (v2 > best) { best = v2; bi = k + 32; } }
    key[gid] = bi * 4096 + s;      // bucket*S + pos (unique -> stable argsort)
}

// ---------------------------------------------------------------------------
// Bitonic argsort of 4096 unique int keys per block (one block per r,b,h).
// ---------------------------------------------------------------------------
__global__ __launch_bounds__(1024) void sort_kernel(
    const int* __restrict__ key, int* __restrict__ sidx)
{
    __shared__ int sk[4096];
    __shared__ int sv[4096];
    const int tid = threadIdx.x;
    const size_t base = (size_t)blockIdx.x * 4096;
    for (int i = tid; i < 4096; i += 1024) { sk[i] = key[base + i]; sv[i] = i; }
    __syncthreads();
    for (int k = 2; k <= 4096; k <<= 1) {
        for (int j = k >> 1; j > 0; j >>= 1) {
            for (int t = tid; t < 4096; t += 1024) {
                int ixj = t ^ j;
                if (ixj > t) {
                    bool up = ((t & k) == 0);
                    int a = sk[t], c = sk[ixj];
                    if ((a > c) == up) {
                        sk[t] = c; sk[ixj] = a;
                        int tv = sv[t]; sv[t] = sv[ixj]; sv[ixj] = tv;
                    }
                }
            }
            __syncthreads();
        }
    }
    for (int i = tid; i < 4096; i += 1024) sidx[base + i] = sv[i];
}

__global__ __launch_bounds__(256) void zero_kernel(float* __restrict__ p, int n)
{
    int i = blockIdx.x * 256 + threadIdx.x;
    if (i < n) p[i] = 0.0f;
}

// ---------------------------------------------------------------------------
// Chunked intra-bucket attention. Block = (chunk c, rbh). 128 threads, 4 waves.
// scores = sqk @ sqk^T / 8, masked to same bucket, softmax, O = P @ sv.
// Output scattered (fused unsort) with atomicAdd * (1/R) into x[b, pos, h*64+d].
// ---------------------------------------------------------------------------
__global__ __launch_bounds__(128) void lsh_attn_kernel(
    const float* __restrict__ qk, const float* __restrict__ v,
    const int* __restrict__ key, const int* __restrict__ sidx,
    float* __restrict__ x)
{
    __shared__ int idxL[64];
    __shared__ int bktL[64];
    __shared__ unsigned short AqP[64 * 64];   // sqk bf16 [row][k]; reused as P
    __shared__ unsigned short Vt[64 * 64];    // sv transposed: [dh][row]
    __shared__ float Sc[64][65];

    const int tid  = threadIdx.x;
    const int lane = tid & 31;
    const int wv   = tid >> 5;                // wave 0..3 -> rows wv*16
    const int c    = blockIdx.x;              // chunk 0..63
    const int rbh  = blockIdx.y;              // 0..255 = ((r*B)+b)*H+h
    const int h    = rbh & 7;
    const int b    = (rbh >> 3) & 7;
    const size_t sortBase = (size_t)rbh * 4096;

    if (tid < 64) {
        int p = sidx[sortBase + c * 64 + tid];
        idxL[tid] = p;
        bktL[tid] = key[sortBase + p] >> 12;  // key = bucket*4096 + pos
    }
    __syncthreads();

    // gather sorted qk rows (bf16) and sv rows (transposed bf16)
    for (int e = tid; e < 64 * 64; e += 128) {
        int i = e >> 6, j = e & 63;
        size_t row = ((size_t)(b * 4096 + idxL[i])) * 512 + h * 64;
        AqP[i * 64 + j] = f2bf(qk[row + j]);
        Vt[j * 64 + i]  = f2bf(v[row + j]);
    }
    __syncthreads();

    // scores: S = sqk @ sqk^T  (AqP serves as both A and transposed-B tile)
    f32x8 accS[4];
#pragma unroll
    for (int nt = 0; nt < 4; ++nt) accS[nt] = zero8();
#pragma unroll
    for (int ks = 0; ks < 64; ks += 32) {
        bf16x16 a = load_frag_A(AqP, 64, wv * 16, ks, lane);
#pragma unroll
        for (int nt = 0; nt < 4; ++nt) {
            bf16x16 bb = load_frag_B(AqP, 64, nt * 16, ks, lane);
            accS[nt] = wmma_bf16(a, bb, accS[nt]);
        }
    }
    const float scale = 0.125f;               // 1/sqrt(Dh)
#pragma unroll
    for (int nt = 0; nt < 4; ++nt) {
        int n = nt * 16 + (lane & 15);
#pragma unroll
        for (int r = 0; r < 8; ++r) {
            int m = wv * 16 + ((lane >> 4) << 3) + r;
            Sc[m][n] = (bktL[m] == bktL[n]) ? accS[nt][r] * scale : -1.0e9f;
        }
    }
    __syncthreads();

    // row softmax, write probs (bf16) back into AqP
    if (tid < 64) {
        float mx = -3.0e38f;
        for (int j = 0; j < 64; ++j) mx = fmaxf(mx, Sc[tid][j]);
        float sum = 0.0f;
        for (int j = 0; j < 64; ++j) sum += __expf(Sc[tid][j] - mx);
        float inv = 1.0f / sum;
        for (int j = 0; j < 64; ++j)
            AqP[tid * 64 + j] = f2bf(__expf(Sc[tid][j] - mx) * inv);
    }
    __syncthreads();

    // O = P @ sv
    f32x8 accO[4];
#pragma unroll
    for (int nt = 0; nt < 4; ++nt) accO[nt] = zero8();
#pragma unroll
    for (int ks = 0; ks < 64; ks += 32) {
        bf16x16 a = load_frag_A(AqP, 64, wv * 16, ks, lane);
#pragma unroll
        for (int nt = 0; nt < 4; ++nt) {
            bf16x16 bb = load_frag_B(Vt, 64, nt * 16, ks, lane);
            accO[nt] = wmma_bf16(a, bb, accO[nt]);
        }
    }

    // scatter-unsort + mean over R rounds (factor 0.25)
#pragma unroll
    for (int nt = 0; nt < 4; ++nt) {
        int n = nt * 16 + (lane & 15);
#pragma unroll
        for (int r = 0; r < 8; ++r) {
            int m = wv * 16 + ((lane >> 4) << 3) + r;
            int p = idxL[m];
            atomicAdd(&x[((size_t)(b * 4096 + p)) * 512 + h * 64 + n],
                      accO[nt][r] * 0.25f);
        }
    }
}

// ---------------------------------------------------------------------------
// Host-side pipeline
// ---------------------------------------------------------------------------
extern "C" void kernel_launch(void* const* d_in, const int* in_sizes, int n_in,
                              void* d_out, int out_size, void* d_ws, size_t ws_size,
                              hipStream_t stream) {
    const float* src = (const float*)d_in[0];   // [8,4096,512]
    const float* Wqk = (const float*)d_in[1];   // [512,512]
    const float* Wv  = (const float*)d_in[2];   // [512,512]
    const float* Wo  = (const float*)d_in[3];   // [512,512]
    const float* rot = (const float*)d_in[4];   // [4,64,32]
    const float* W1  = (const float*)d_in[5];   // [512,2048]
    const float* b1  = (const float*)d_in[6];   // [2048]
    const float* W2  = (const float*)d_in[7];   // [2048,512]
    const float* b2  = (const float*)d_in[8];   // [512]
    float* out = (float*)d_out;                 // [8,4096,512]

    const int M = 32768;                        // B*S
    const size_t SZ_BSD = (size_t)M * 512 * 4;  // 64 MB
    const size_t SZ_KEY = (size_t)4 * 8 * 8 * 4096 * 4;  // 4 MB

    char* ws = (char*)d_ws;
    float* qk   = (float*)(ws);                               // 64 MB
    float* vv   = (float*)(ws + SZ_BSD);                      // 64 MB
    int*   keyB = (int*)  (ws + 2 * SZ_BSD);                  // 4 MB
    int*   sidx = (int*)  (ws + 2 * SZ_BSD + SZ_KEY);         // 4 MB
    float* xatt = (float*)(ws + 2 * SZ_BSD + 2 * SZ_KEY);     // 64 MB
    float* yatt = (float*)(ws + 3 * SZ_BSD + 2 * SZ_KEY);     // 64 MB
    // FFN hidden (bf16, 128 MB) aliases qk+vv which are dead by then
    unsigned short* hbuf = (unsigned short*)ws;

    // 1-2. projections: qk = src@Wqk, v = src@Wv
    gemm_dbuf_kernel<float><<<dim3(M / 128, 4), 256, 0, stream>>>(
        src, Wqk, nullptr, qk, 0, 0, M, 512, 512);
    gemm_dbuf_kernel<float><<<dim3(M / 128, 4), 256, 0, stream>>>(
        src, Wv, nullptr, vv, 0, 0, M, 512, 512);

    // 3. LSH hashing -> sort keys
    hash_kernel<<<4194304 / 256, 256, 0, stream>>>(qk, rot, keyB);

    // 4. per-(r,b,h) bitonic argsort
    sort_kernel<<<256, 1024, 0, stream>>>(keyB, sidx);

    // 5. zero the attention accumulator
    zero_kernel<<<(M * 512) / 256, 256, 0, stream>>>(xatt, M * 512);

    // 6. chunked intra-bucket attention with fused unsort + round-mean
    lsh_attn_kernel<<<dim3(64, 256), 128, 0, stream>>>(qk, vv, keyB, sidx, xatt);

    // 7. output projection: y = x @ Wo
    gemm_dbuf_kernel<float><<<dim3(M / 128, 4), 256, 0, stream>>>(
        xatt, Wo, nullptr, yatt, 0, 0, M, 512, 512);

    // 8. FFN up: h = relu(y @ W1 + b1), stored bf16
    gemm_dbuf_kernel<float><<<dim3(M / 128, 16), 256, 0, stream>>>(
        yatt, W1, b1, hbuf, 1, 1, M, 2048, 512);

    // 9. FFN down: out = h @ W2 + b2
    gemm_dbuf_kernel<unsigned short><<<dim3(M / 128, 4), 256, 0, stream>>>(
        hbuf, W2, b2, out, 0, 0, M, 512, 2048);
}
